// Painn_90761248899583
// MI455X (gfx1250) — compile-verified, matrix operand order checked by reading, not run on hardware
//
#include <hip/hip_runtime.h>
#include <math.h>

// ---------------------------------------------------------------------------
// Types / constants
// ---------------------------------------------------------------------------
typedef __bf16 bf16_t;
typedef bf16_t v16bf __attribute__((ext_vector_type(16)));
typedef float  v8f   __attribute__((ext_vector_type(8)));

constexpr int   kF      = 128;   // feature dim
constexpr int   kR      = 20;    // radial basis count
constexpr int   kL      = 3;     // layers
constexpr float kCutoff = 5.0f;
constexpr float kEps    = 1e-15f;
constexpr float kPi     = 3.14159265358979323846f;

enum { ACT_NONE = 0, ACT_SILU = 1, ACT_RELU = 2 };

// ---------------------------------------------------------------------------
// Weight pre-swizzle: pack B[K x Nc] (f32, row-major) into per-lane WMMA
// fragment order so each lane reads its 16 bf16 fragment elements as one
// contiguous 32-byte chunk:
//   out[((ktile*(Nc/16) + ntile)*32 + lane)*16 + e]
//     = (bf16) B[k, c],  k = ktile*32 + (e>>3)*16 + (lane>>4)*8 + (e&7)
//                        c = ntile*16 + (lane&15)
// ---------------------------------------------------------------------------
__global__ void k_pack_b(const float* __restrict__ W, bf16_t* __restrict__ out,
                         int K, int Nc) {
  const int i = blockIdx.x * blockDim.x + threadIdx.x;
  if (i < K * Nc) {
    const int r    = i & 511;        // position within one 16x32 tile (512 el)
    const int tile = i >> 9;
    const int lane = r >> 4;
    const int e    = r & 15;
    const int ntiles = Nc >> 4;
    const int ktile  = tile / ntiles;
    const int ntile  = tile - ktile * ntiles;
    const int k = ktile * 32 + ((e >> 3) * 16) + ((lane >> 4) * 8) + (e & 7);
    const int c = ntile * 16 + (lane & 15);
    out[i] = (bf16_t)W[(size_t)k * Nc + c];   // fptrunc, round-to-nearest-even
  }
}

// ---------------------------------------------------------------------------
// GEMM: C[M x Nc] = act(A[M x K](f32) * Bp(pre-swizzled bf16) + bias[Nc])
// block = 128 threads (4 waves); wave w owns NT adjacent 16x16 output tiles
// starting at column blockIdx.y*(64*NT) + w*(16*NT). One A fragment feeds
// 2*NT WMMAs per K-step of 64 (2*NT independent accumulator chains).
// Requires M%16==0, K%64==0, Nc%(64*NT)==0.
// ---------------------------------------------------------------------------
template <int ACT, int NT>
__global__ __launch_bounds__(128)
void gemm_wmma_bf16(const float* __restrict__ A, const bf16_t* __restrict__ Bp,
                    const float* __restrict__ bias, float* __restrict__ C,
                    int M, int K, int Nc) {
  const int lane = threadIdx.x & 31;
  const int wave = threadIdx.x >> 5;
  const int hh   = lane >> 4;   // half-wave select (0/1)
  const int li   = lane & 15;
  const int m0   = blockIdx.x * 16;
  const int n0   = (blockIdx.y * 4 + wave) * (16 * NT);
  const int ntiles = Nc >> 4;

  // A: lane = row, element e -> K = k0 + (e>>3)*16 + hh*8 + (e&7)
  const float* arow = A + (size_t)(m0 + li) * K + hh * 8;
  // B: pre-swizzled; one 32B fragment per lane per 16x32 tile
  const bf16_t* bbase = Bp + ((size_t)(n0 >> 4) * 32 + lane) * 16;
  const size_t  bstep = (size_t)ntiles * 512;   // elements per k-tile of 32

  v8f acc0[NT], acc1[NT];
#pragma unroll
  for (int nt = 0; nt < NT; ++nt) {
    acc0[nt] = (v8f){0.f, 0.f, 0.f, 0.f, 0.f, 0.f, 0.f, 0.f};
    acc1[nt] = (v8f){0.f, 0.f, 0.f, 0.f, 0.f, 0.f, 0.f, 0.f};
  }

  for (int k0 = 0; k0 < K; k0 += 64) {
    const v8f a00 = *(const v8f*)(arow + k0);        // K: k0+hh*8    ..+7
    const v8f a01 = *(const v8f*)(arow + k0 + 16);   // K: k0+16+hh*8 ..+7
    const v8f a10 = *(const v8f*)(arow + k0 + 32);
    const v8f a11 = *(const v8f*)(arow + k0 + 48);
    v16bf af0, af1;
#pragma unroll
    for (int e = 0; e < 8; ++e) {
      af0[e]     = (bf16_t)a00[e];
      af0[8 + e] = (bf16_t)a01[e];
      af1[e]     = (bf16_t)a10[e];
      af1[8 + e] = (bf16_t)a11[e];
    }
    const bf16_t* bk0 = bbase + (size_t)(k0 >> 5) * bstep;
#pragma unroll
    for (int nt = 0; nt < NT; ++nt) {
      const v16bf bf0 = *(const v16bf*)(bk0 + nt * 512);
      const v16bf bf1 = *(const v16bf*)(bk0 + bstep + nt * 512);
      acc0[nt] = __builtin_amdgcn_wmma_f32_16x16x32_bf16(
          false, af0, false, bf0, (short)0, acc0[nt], false, false);
      acc1[nt] = __builtin_amdgcn_wmma_f32_16x16x32_bf16(
          false, af1, false, bf1, (short)0, acc1[nt], false, false);
    }
  }

#pragma unroll
  for (int nt = 0; nt < NT; ++nt) {
    const int   col  = n0 + nt * 16 + li;
    const float bval = bias[col];
    float* crow = C + (size_t)(m0 + hh * 8) * Nc + col;  // D: VGPR r -> M=r+hh*8
#pragma unroll
    for (int r = 0; r < 8; ++r) {
      float x = acc0[nt][r] + acc1[nt][r] + bval;
      if (ACT == ACT_SILU)      x = x * __builtin_amdgcn_rcpf(1.0f + __expf(-x));
      else if (ACT == ACT_RELU) x = fmaxf(x, 0.0f);
      crow[(size_t)r * Nc] = x;
    }
  }
}

// ---------------------------------------------------------------------------
// Edge message kernel. 128 threads = 128 feature channels; grid-stride over
// edges. Per edge, channel c:
//   w_k = fcut * (bd[kF*k+c] + sum_r rbf[r]*Wd[r][kF*k+c])   k=0,1,2
//   s0/s1/s2 = phi[j][c + kF*k] * w_k
//   atomic: s[i][c]     += s1
//           dv[d][i][c] += s2*unit[d] + s0*v_old[d][j][c]
// ---------------------------------------------------------------------------
__global__ __launch_bounds__(128)
void edge_message(const float* __restrict__ xyz, const int* __restrict__ nbrs,
                  const float* __restrict__ phi,   // N x 3F
                  const float* __restrict__ Wd,    // R x 3F
                  const float* __restrict__ bd,    // 3F
                  const float* __restrict__ v_old, // 3 planes of N*F
                  float* __restrict__ s_acc,       // N x F (atomic target)
                  float* __restrict__ dv,          // 3 planes N*F (atomic)
                  int n_edges, int n_atoms) {
  __shared__ float sWd[kR * 3 * kF];   // 30720 B
  __shared__ float sbd[3 * kF];
  __shared__ float srbf[kR];
  __shared__ float sunit[3];
  __shared__ float sfc[1];

  const int t = threadIdx.x;
  for (int i = t; i < kR * 3 * kF; i += 128) sWd[i] = Wd[i];
  for (int i = t; i < 3 * kF; i += 128)      sbd[i] = bd[i];
  __syncthreads();

  const size_t plane = (size_t)n_atoms * kF;

  for (int e = blockIdx.x; e < n_edges; e += gridDim.x) {
    const int ai = nbrs[2 * e + 0];
    const int aj = nbrs[2 * e + 1];
    if (t < 24) {
      const float dx = xyz[3 * aj + 0] - xyz[3 * ai + 0];
      const float dy = xyz[3 * aj + 1] - xyz[3 * ai + 1];
      const float dz = xyz[3 * aj + 2] - xyz[3 * ai + 2];
      const float dist = sqrtf(dx * dx + dy * dy + dz * dz + kEps);
      const float rinv = __builtin_amdgcn_rcpf(dist);
      if (t < kR) {
        srbf[t] = __sinf((float)(t + 1) * (kPi / kCutoff) * dist) * rinv;
      } else if (t == 20) { sunit[0] = dx * rinv; }
      else if (t == 21)   { sunit[1] = dy * rinv; }
      else if (t == 22)   { sunit[2] = dz * rinv; }
      else {
        sfc[0] = (dist < kCutoff)
                     ? 0.5f * (__cosf((kPi / kCutoff) * dist) + 1.0f)
                     : 0.0f;
      }
    }
    __syncthreads();

    // Prefetch next edge's gather targets while we run the RBF filter.
    const int en = e + gridDim.x;
    if (en < n_edges) {
      const int ajn = nbrs[2 * en + 1];
      if (t < 12) {
        __builtin_prefetch(phi + (size_t)ajn * (3 * kF) + t * 32, 0, 1);
      } else if (t < 24) {
        const int d = (t - 12) >> 2, q = (t - 12) & 3;
        __builtin_prefetch(v_old + (size_t)ajn * kF + (size_t)d * plane + q * 32,
                           0, 1);
      }
    }

    float w0 = sbd[t], w1 = sbd[kF + t], w2 = sbd[2 * kF + t];
#pragma unroll
    for (int r = 0; r < kR; ++r) {
      const float rb = srbf[r];
      const float* wrow = &sWd[r * 3 * kF];
      w0 += rb * wrow[t];
      w1 += rb * wrow[kF + t];
      w2 += rb * wrow[2 * kF + t];
    }
    const float fc = sfc[0];
    w0 *= fc; w1 *= fc; w2 *= fc;

    const float* pj = phi + (size_t)aj * (3 * kF);
    const float s0 = pj[t] * w0;
    const float s1 = pj[kF + t] * w1;
    const float s2 = pj[2 * kF + t] * w2;

    const size_t ii = (size_t)ai * kF + t;
    const size_t jj = (size_t)aj * kF + t;
    atomicAdd(&s_acc[ii], s1);
    atomicAdd(&dv[ii],             s2 * sunit[0] + s0 * v_old[jj]);
    atomicAdd(&dv[ii + plane],     s2 * sunit[1] + s0 * v_old[jj + plane]);
    atomicAdd(&dv[ii + 2 * plane], s2 * sunit[2] + s0 * v_old[jj + 2 * plane]);
    __syncthreads();
  }
}

// ---------------------------------------------------------------------------
// Small elementwise kernels
// ---------------------------------------------------------------------------
__global__ void k_zero(float* __restrict__ p, int n) {
  const int i = blockIdx.x * blockDim.x + threadIdx.x;
  if (i < n) p[i] = 0.0f;
}

__global__ void k_add(float* __restrict__ a, const float* __restrict__ b, int n) {
  const int i = blockIdx.x * blockDim.x + threadIdx.x;
  if (i < n) a[i] += b[i];
}

__global__ void k_embed(const int* __restrict__ z, const float* __restrict__ embed,
                        float* __restrict__ s, int n_atoms) {
  const int i = blockIdx.x * blockDim.x + threadIdx.x;
  if (i < n_atoms * kF) {
    const int n = i / kF, f = i % kF;
    s[i] = embed[(size_t)z[n] * kF + f];
  }
}

__global__ void k_vnorm_stack(const float* __restrict__ s, const float* __restrict__ vv,
                              float* __restrict__ stack, int n_atoms) {
  const int i = blockIdx.x * blockDim.x + threadIdx.x;
  if (i < n_atoms * kF) {
    const int n = i / kF, g = i % kF;
    const size_t plane = (size_t)n_atoms * kF;
    const float a = vv[i], b = vv[i + plane], c = vv[i + 2 * plane];
    stack[(size_t)n * 2 * kF + g]      = s[i];
    stack[(size_t)n * 2 * kF + kF + g] = sqrtf(a * a + b * b + c * c + kEps);
  }
}

__global__ void k_update(float* __restrict__ s, float* __restrict__ v,
                         const float* __restrict__ uv, const float* __restrict__ vv,
                         const float* __restrict__ split, int n_atoms) {
  const int i = blockIdx.x * blockDim.x + threadIdx.x;
  if (i < n_atoms * kF) {
    const int n = i / kF, g = i % kF;
    const size_t plane = (size_t)n_atoms * kF;
    const size_t sb = (size_t)n * 3 * kF;
    const float a_vv = split[sb + g];
    const float a_sv = split[sb + kF + g];
    const float a_ss = split[sb + 2 * kF + g];
    float dot = 0.0f;
#pragma unroll
    for (int d = 0; d < 3; ++d) {
      const float u = uv[i + d * plane];
      dot += u * vv[i + d * plane];
      v[i + d * plane] += u * a_vv;
    }
    s[i] += dot * a_sv + a_ss;
  }
}

__global__ __launch_bounds__(128)
void k_final(const float* __restrict__ h, const float* __restrict__ Wo,
             const float* __restrict__ bo, float* __restrict__ out, int n_atoms) {
  __shared__ float red[128];
  const int n = blockIdx.x;
  const int t = threadIdx.x;
  red[t] = h[(size_t)n * kF + t] * Wo[t];
  __syncthreads();
  for (int off = 64; off > 0; off >>= 1) {
    if (t < off) red[t] += red[t + off];
    __syncthreads();
  }
  if (t == 0) out[n] = red[0] + bo[0];
}

// ---------------------------------------------------------------------------
// Host orchestration
// ---------------------------------------------------------------------------
static inline int cdiv(int a, int b) { return (a + b - 1) / b; }

static void launch_gemm(int act, const float* A, const bf16_t* B, const float* bias,
                        float* C, int M, int K, int Nc, hipStream_t stream) {
  dim3 block(128);
  if (Nc % 128 == 0) {
    dim3 grid(M / 16, Nc / 128);
    if (act == ACT_SILU)
      gemm_wmma_bf16<ACT_SILU, 2><<<grid, block, 0, stream>>>(A, B, bias, C, M, K, Nc);
    else if (act == ACT_RELU)
      gemm_wmma_bf16<ACT_RELU, 2><<<grid, block, 0, stream>>>(A, B, bias, C, M, K, Nc);
    else
      gemm_wmma_bf16<ACT_NONE, 2><<<grid, block, 0, stream>>>(A, B, bias, C, M, K, Nc);
  } else {
    dim3 grid(M / 16, Nc / 64);
    if (act == ACT_SILU)
      gemm_wmma_bf16<ACT_SILU, 1><<<grid, block, 0, stream>>>(A, B, bias, C, M, K, Nc);
    else if (act == ACT_RELU)
      gemm_wmma_bf16<ACT_RELU, 1><<<grid, block, 0, stream>>>(A, B, bias, C, M, K, Nc);
    else
      gemm_wmma_bf16<ACT_NONE, 1><<<grid, block, 0, stream>>>(A, B, bias, C, M, K, Nc);
  }
}

extern "C" void kernel_launch(void* const* d_in, const int* in_sizes, int n_in,
                              void* d_out, int out_size, void* d_ws, size_t ws_size,
                              hipStream_t stream) {
  const float* xyz     = (const float*)d_in[0];
  const int*   z       = (const int*)  d_in[1];
  const int*   nbrs    = (const int*)  d_in[2];
  const float* embed   = (const float*)d_in[3];
  const float* msg_W1  = (const float*)d_in[4];
  const float* msg_b1  = (const float*)d_in[5];
  const float* msg_W2  = (const float*)d_in[6];
  const float* msg_b2  = (const float*)d_in[7];
  const float* msg_Wd  = (const float*)d_in[8];
  const float* msg_bd  = (const float*)d_in[9];
  const float* upd_U   = (const float*)d_in[10];
  const float* upd_V   = (const float*)d_in[11];
  const float* upd_Ws1 = (const float*)d_in[12];
  const float* upd_bs1 = (const float*)d_in[13];
  const float* upd_Ws2 = (const float*)d_in[14];
  const float* upd_bs2 = (const float*)d_in[15];
  const float* ro_W1   = (const float*)d_in[16];
  const float* ro_b1   = (const float*)d_in[17];
  const float* ro_W2   = (const float*)d_in[18];
  const float* ro_b2   = (const float*)d_in[19];
  const float* fc_W1   = (const float*)d_in[20];
  const float* fc_b1   = (const float*)d_in[21];
  const float* fc_W2   = (const float*)d_in[22];
  const float* fc_b2   = (const float*)d_in[23];
  const float* fc_Wo   = (const float*)d_in[24];
  const float* fc_bo   = (const float*)d_in[25];

  const int N = in_sizes[0] / 3;   // 20000
  const int E = in_sizes[2] / 2;   // 320000
  const size_t NF = (size_t)N * kF;
  const int iNF  = (int)NF;
  const int i3NF = (int)(3 * NF);

  // fp32 workspace layout (buffers reused across phases; 14*NF floats total)
  float* s      = (float*)d_ws;       // NF   persistent scalar features
  float* v      = s + NF;             // 3NF  persistent vectors, planar [d][n][f]
  float* bufA   = v + 3 * NF;         // 3NF  dv -> u_v -> readout h3/h4
  float* bufB   = bufA + 3 * NF;      // 3NF  phi -> v_v
  float* bufC   = bufB + 3 * NF;      // 3NF  stack -> split
  float* phipre = bufC + 3 * NF;      // NF   gemm temporaries / readout h1,h2
  float* zbias  = phipre + NF;        // 384  zero bias for einsum GEMMs
  bf16_t* bw    = (bf16_t*)(zbias + 384);

  // bf16 pre-swizzled weight region
  bf16_t* bW1   = bw;
  bf16_t* bW2   = bW1   + (size_t)kL * kF * kF;
  bf16_t* bU    = bW2   + (size_t)kL * kF * 3 * kF;
  bf16_t* bV    = bU    + (size_t)kL * kF * kF;
  bf16_t* bWs1  = bV    + (size_t)kL * kF * kF;
  bf16_t* bWs2  = bWs1  + (size_t)kL * 2 * kF * kF;
  bf16_t* broW1 = bWs2  + (size_t)kL * kF * 3 * kF;
  bf16_t* broW2 = broW1 + (size_t)kF * (kF / 2);
  bf16_t* bfcW1 = broW2 + (size_t)(kF / 2) * 64;
  bf16_t* bfcW2 = bfcW1 + (size_t)64 * kF;

  const int TB = 256;
  auto pack = [&](const float* W, bf16_t* out, int K, int Nc) {
    const int n = K * Nc;
    k_pack_b<<<cdiv(n, TB), TB, 0, stream>>>(W, out, K, Nc);
  };
  for (int l = 0; l < kL; ++l) {
    pack(msg_W1  + (size_t)l * kF * kF,     bW1  + (size_t)l * kF * kF,     kF,     kF);
    pack(msg_W2  + (size_t)l * kF * 3 * kF, bW2  + (size_t)l * kF * 3 * kF, kF, 3 * kF);
    pack(upd_U   + (size_t)l * kF * kF,     bU   + (size_t)l * kF * kF,     kF,     kF);
    pack(upd_V   + (size_t)l * kF * kF,     bV   + (size_t)l * kF * kF,     kF,     kF);
    pack(upd_Ws1 + (size_t)l * 2 * kF * kF, bWs1 + (size_t)l * 2 * kF * kF, 2 * kF, kF);
    pack(upd_Ws2 + (size_t)l * kF * 3 * kF, bWs2 + (size_t)l * kF * 3 * kF, kF, 3 * kF);
  }
  pack(ro_W1, broW1, kF, 64);
  pack(ro_W2, broW2, 64, 64);
  pack(fc_W1, bfcW1, 64, kF);
  pack(fc_W2, bfcW2, kF, kF);

  // init: s = embed[z], v = 0, zbias = 0
  k_embed<<<cdiv(iNF, TB), TB, 0, stream>>>(z, embed, s, N);
  k_zero<<<cdiv(i3NF, TB), TB, 0, stream>>>(v, i3NF);
  k_zero<<<1, 384, 0, stream>>>(zbias, 384);

  for (int l = 0; l < kL; ++l) {
    // phi = silu(s @ W1 + b1) @ W2 + b2
    launch_gemm(ACT_SILU, s, bW1 + (size_t)l * kF * kF,
                msg_b1 + (size_t)l * kF, phipre, N, kF, kF, stream);
    launch_gemm(ACT_NONE, phipre, bW2 + (size_t)l * kF * 3 * kF,
                msg_b2 + (size_t)l * 3 * kF, bufB /*phi*/, N, kF, 3 * kF, stream);

    // edge scatter: s += seg(s1); dv = seg(s2*unit + s0*v[j])
    k_zero<<<cdiv(i3NF, TB), TB, 0, stream>>>(bufA /*dv*/, i3NF);
    edge_message<<<2048, 128, 0, stream>>>(
        xyz, nbrs, bufB /*phi*/, msg_Wd + (size_t)l * kR * 3 * kF,
        msg_bd + (size_t)l * 3 * kF, v, s, bufA /*dv*/, E, N);
    k_add<<<cdiv(i3NF, TB), TB, 0, stream>>>(v, bufA /*dv*/, i3NF);

    // u_v = v @ U, v_v = v @ V  (planar v treated as (3N x F) matrix)
    launch_gemm(ACT_NONE, v, bU + (size_t)l * kF * kF, zbias,
                bufA /*u_v*/, 3 * N, kF, kF, stream);
    launch_gemm(ACT_NONE, v, bV + (size_t)l * kF * kF, zbias,
                bufB /*v_v*/, 3 * N, kF, kF, stream);

    // stack = [s, ||v_v||]; split = silu(stack@Ws1+bs1)@Ws2+bs2
    k_vnorm_stack<<<cdiv(iNF, TB), TB, 0, stream>>>(s, bufB /*v_v*/, bufC, N);
    launch_gemm(ACT_SILU, bufC /*stack*/, bWs1 + (size_t)l * 2 * kF * kF,
                upd_bs1 + (size_t)l * kF, phipre, N, 2 * kF, kF, stream);
    launch_gemm(ACT_NONE, phipre, bWs2 + (size_t)l * kF * 3 * kF,
                upd_bs2 + (size_t)l * 3 * kF, bufC /*split*/, N, kF, 3 * kF, stream);

    // v += u_v*a_vv ; s += dot(u_v,v_v)*a_sv + a_ss
    k_update<<<cdiv(iNF, TB), TB, 0, stream>>>(s, v, bufA /*u_v*/, bufB /*v_v*/,
                                               bufC /*split*/, N);
  }

  // Readout MLP
  float* h1 = phipre;                       // N x 64
  float* h2 = phipre + (size_t)N * 64;      // N x 64
  float* h3 = bufA;                         // N x 128
  float* h4 = bufA + NF;                    // N x 128
  launch_gemm(ACT_SILU, s,  broW1, ro_b1, h1, N, kF, 64, stream);
  launch_gemm(ACT_NONE, h1, broW2, ro_b2, h2, N, 64, 64, stream);
  launch_gemm(ACT_RELU, h2, bfcW1, fc_b1, h3, N, 64, kF, stream);
  launch_gemm(ACT_RELU, h3, bfcW2, fc_b2, h4, N, kF, kF, stream);
  k_final<<<N, 128, 0, stream>>>(h4, fc_Wo, fc_bo, (float*)d_out, N);
}
